// Main_Net_79852031967519
// MI455X (gfx1250) — matrix operand order repaired and checked by reading, not compiled, not run on hardware
//
#include <hip/hip_runtime.h>
#include <hip/hip_bf16.h>
#include <stdint.h>

// ---- types for CDNA5 WMMA ----
typedef __bf16 bf16_t;
typedef bf16_t   v16bf __attribute__((ext_vector_type(16)));
typedef float    v8f   __attribute__((ext_vector_type(8)));
typedef uint32_t v4u   __attribute__((ext_vector_type(4)));

#define B_TOTAL 16384
#define DIN     1024
#define HID     512
#define COMP    64
#define ROWS1   128   // rows/WG for the big branch GEMM
#define ROWS2   64    // rows/WG for compress + kron kernels
#define KC      32
#define LDSS    40    // padded bf16 row stride (80B) -> conflict-free b128 reads

// A fragment: 16x32 bf16, row-major LDS tile [16 rows][LDSS]
__device__ __forceinline__ v16bf frag_a(const bf16_t* t) {
  const int lane = threadIdx.x & 31;
  const int m = lane & 15, half = lane >> 4;
  const bf16_t* p = t + m * LDSS + half * 8;
  union { v16bf v; v4u q[2]; } u;
  u.q[0] = *(const v4u*)(p);        // K = half*8 + 0..7
  u.q[1] = *(const v4u*)(p + 16);   // K = 16 + half*8 + 0..7
  return u.v;
}

// B fragment: 32x16 bf16 (KxN), LDS tile stored TRANSPOSED [16 n][LDSS k]
__device__ __forceinline__ v16bf frag_b(const bf16_t* t) {
  const int lane = threadIdx.x & 31;
  const int n = lane & 15, half = lane >> 4;
  const bf16_t* p = t + n * LDSS + half * 8;
  union { v16bf v; v4u q[2]; } u;
  u.q[0] = *(const v4u*)(p);
  u.q[1] = *(const v4u*)(p + 16);
  return u.v;
}

__device__ __forceinline__ v8f wmma_bf16(v16bf a, v16bf b, v8f c) {
  return __builtin_amdgcn_wmma_f32_16x16x32_bf16(false, a, false, b, (short)0, c,
                                                 false, false);
}

union pk4 { bf16_t h[4]; uint2 u; };

// ---- Prep: fp32 W[K][N] -> bf16 WT[N][K], LDS-tiled for coalescing ----
__global__ __launch_bounds__(256)
void k_cvt_t(const float* __restrict__ S, bf16_t* __restrict__ D, int K, int N) {
  __shared__ bf16_t t[64 * 72];
  const int tid = threadIdx.x;
  const int k0 = blockIdx.x * 64, n0 = blockIdx.y * 64;
  for (int i = tid; i < 64 * 64; i += 256) {
    int r = i >> 6, c = i & 63;             // r=k_local, c=n_local
    t[c * 72 + r] = (bf16_t)S[(size_t)(k0 + r) * N + n0 + c];
  }
  __syncthreads();
  for (int i = tid; i < 64 * 16; i += 256) {
    int r = i >> 4, c4 = (i & 15) * 4;      // r=n_local, c4=k_local
    pk4 w;
    w.h[0] = t[r * 72 + c4 + 0];
    w.h[1] = t[r * 72 + c4 + 1];
    w.h[2] = t[r * 72 + c4 + 2];
    w.h[3] = t[r * 72 + c4 + 3];
    *(uint2*)&D[(size_t)(n0 + r) * K + k0 + c4] = w.u;
  }
}

// ---------------- Kernel 1: h = relu(X @ Wb + bb), bf16 out ----------------
// WT = pre-transposed bf16 weights [HID][DIN]
__global__ __launch_bounds__(256)
void k_branch(const float* __restrict__ X1, const bf16_t* __restrict__ WT1,
              const float* __restrict__ b1, bf16_t* __restrict__ H1,
              const float* __restrict__ X2, const bf16_t* __restrict__ WT2,
              const float* __restrict__ b2, bf16_t* __restrict__ H2)
{
  const float*  X    = blockIdx.y ? X2 : X1;
  const bf16_t* WT   = blockIdx.y ? WT2 : WT1;
  const float*  bias = blockIdx.y ? b2 : b1;
  bf16_t*       H    = blockIdx.y ? H2 : H1;

  __shared__ bf16_t sA[ROWS1 * LDSS];   // 10 KB
  __shared__ bf16_t sBT[HID * LDSS];    // 40 KB, [n][k]

  const int tid  = threadIdx.x;
  const int wave = tid >> 5;
  const int row0 = blockIdx.x * ROWS1;

  v8f acc[8][4];
  #pragma unroll
  for (int mt = 0; mt < 8; ++mt)
    #pragma unroll
    for (int nt = 0; nt < 4; ++nt)
      #pragma unroll
      for (int r = 0; r < 8; ++r) acc[mt][nt][r] = 0.f;

  for (int kc = 0; kc < DIN / KC; ++kc) {
    if (kc + 1 < DIN / KC)
      __builtin_prefetch(WT + (size_t)(kc + 1) * KC, 0, 1);
    // stage X tile (128x32 fp32 -> bf16), float4 vectorized
    for (int i = tid; i < ROWS1 * KC / 4; i += 256) {
      int r = i >> 3, c4 = (i & 7) * 4;
      float4 v = *(const float4*)&X[(size_t)(row0 + r) * DIN + kc * KC + c4];
      pk4 w;
      w.h[0] = (bf16_t)v.x; w.h[1] = (bf16_t)v.y;
      w.h[2] = (bf16_t)v.z; w.h[3] = (bf16_t)v.w;
      *(uint2*)&sA[r * LDSS + c4] = w.u;
    }
    // stage WT tile: pure b128 copy of contiguous k-runs
    for (int i = tid; i < HID * 4; i += 256) {
      int n = i >> 2, seg = (i & 3) * 8;
      *(v4u*)&sBT[n * LDSS + seg] =
          *(const v4u*)&WT[(size_t)n * DIN + kc * KC + seg];
    }
    __syncthreads();

    v16bf a[8];
    #pragma unroll
    for (int mt = 0; mt < 8; ++mt) a[mt] = frag_a(sA + mt * 16 * LDSS);
    #pragma unroll
    for (int nt = 0; nt < 4; ++nt) {
      v16bf b = frag_b(sBT + (wave * 64 + nt * 16) * LDSS);
      #pragma unroll
      for (int mt = 0; mt < 8; ++mt)
        acc[mt][nt] = wmma_bf16(a[mt], b, acc[mt][nt]);
    }
    __syncthreads();
  }

  const int lane = tid & 31, lm = lane & 15, lh = lane >> 4;
  #pragma unroll
  for (int nt = 0; nt < 4; ++nt) {
    const int col = wave * 64 + nt * 16 + lm;
    const float bv = bias[col];
    #pragma unroll
    for (int mt = 0; mt < 8; ++mt)
      #pragma unroll
      for (int r = 0; r < 8; ++r) {
        const int row = row0 + mt * 16 + lh * 8 + r;
        float v = acc[mt][nt][r] + bv;
        H[(size_t)row * HID + col] = (bf16_t)(v > 0.f ? v : 0.f);
      }
  }
}

// ---------------- Kernel 2: f = h @ Wc + bc, fp32 out ----------------
// WcT = pre-transposed bf16 [COMP][HID]
__global__ __launch_bounds__(256)
void k_compress(const bf16_t* __restrict__ Ha, const bf16_t* __restrict__ WTa,
                const float* __restrict__ bca, float* __restrict__ Fa,
                const bf16_t* __restrict__ Hb, const bf16_t* __restrict__ WTb,
                const float* __restrict__ bcb, float* __restrict__ Fb)
{
  const bf16_t* H  = blockIdx.y ? Hb : Ha;
  const bf16_t* WT = blockIdx.y ? WTb : WTa;
  const float*  bc = blockIdx.y ? bcb : bca;
  float*        F  = blockIdx.y ? Fb : Fa;

  __shared__ bf16_t sA[ROWS2 * LDSS];
  __shared__ bf16_t sBT[COMP * LDSS];

  const int tid  = threadIdx.x;
  const int wave = tid >> 5;
  const int row0 = blockIdx.x * ROWS2;
  const int mt  = wave & 3;
  const int nt0 = (wave >> 2) * 2;

  v8f acc[2];
  #pragma unroll
  for (int j = 0; j < 2; ++j)
    #pragma unroll
    for (int r = 0; r < 8; ++r) acc[j][r] = 0.f;

  for (int kc = 0; kc < HID / KC; ++kc) {
    { // h tile: b128 copy (64 rows x 4 segs = 256 transfers, 1/thread)
      int r = tid >> 2, seg = (tid & 3) * 8;
      *(v4u*)&sA[r * LDSS + seg] =
          *(const v4u*)&H[(size_t)(row0 + r) * HID + kc * KC + seg];
    }
    { // WcT tile: b128 copy
      int n = tid >> 2, seg = (tid & 3) * 8;
      *(v4u*)&sBT[n * LDSS + seg] =
          *(const v4u*)&WT[(size_t)n * HID + kc * KC + seg];
    }
    __syncthreads();
    v16bf a = frag_a(sA + mt * 16 * LDSS);
    #pragma unroll
    for (int j = 0; j < 2; ++j) {
      v16bf b = frag_b(sBT + (nt0 + j) * 16 * LDSS);
      acc[j] = wmma_bf16(a, b, acc[j]);
    }
    __syncthreads();
  }

  const int lane = tid & 31, lm = lane & 15, lh = lane >> 4;
  #pragma unroll
  for (int j = 0; j < 2; ++j) {
    const int col = (nt0 + j) * 16 + lm;
    const float bv = bc[col];
    #pragma unroll
    for (int r = 0; r < 8; ++r) {
      const int row = row0 + mt * 16 + lh * 8 + r;
      F[(size_t)row * COMP + col] = acc[j][r] + bv;
    }
  }
}

// ------- Kernel 3: kron(f2,f1) @ Wf1 + bf1, then @ Wf2 + bf2 -------
// kron flat k = p*64+q -> A[b][k] = f2[b][p]*f1[b][q]
// 32-wide chunk kc: p = kc>>1 (constant), q = (kc&1)*32 + i
// Wf1T = pre-transposed bf16 [COMP][4096]
__global__ __launch_bounds__(256)
void k_kron_head(const float* __restrict__ F1, const float* __restrict__ F2,
                 const bf16_t* __restrict__ Wf1T, const float* __restrict__ bf1,
                 const float* __restrict__ Wf2, const float* __restrict__ bf2,
                 float* __restrict__ Out)
{
  __shared__ float  sF1[ROWS2 * COMP];   // 16 KB
  __shared__ float  sF2[ROWS2 * COMP];   // 16 KB
  __shared__ bf16_t sA[ROWS2 * LDSS];    // 5 KB
  __shared__ bf16_t sBT[COMP * LDSS];    // 5 KB
  __shared__ float  sH[ROWS2 * 66];      // 16.9 KB

  const int tid  = threadIdx.x;
  const int wave = tid >> 5;
  const int row0 = blockIdx.x * ROWS2;
  const int mt  = wave & 3;
  const int nt0 = (wave >> 2) * 2;

  for (int i = tid; i < ROWS2 * COMP / 4; i += 256) {
    int r = i >> 4, c4 = (i & 15) * 4;
    *(float4*)&sF1[r * COMP + c4] =
        *(const float4*)&F1[(size_t)(row0 + r) * COMP + c4];
    *(float4*)&sF2[r * COMP + c4] =
        *(const float4*)&F2[(size_t)(row0 + r) * COMP + c4];
  }
  __syncthreads();

  v8f acc[2];
  #pragma unroll
  for (int j = 0; j < 2; ++j)
    #pragma unroll
    for (int r = 0; r < 8; ++r) acc[j][r] = 0.f;

  for (int kc = 0; kc < (COMP * COMP) / KC; ++kc) {   // 128 chunks
    const int p  = kc >> 1;
    const int qb = (kc & 1) * KC;
    // generate kron A tile (fp32 product -> packed bf16), vectorized x4
    for (int i = tid; i < ROWS2 * KC / 4; i += 256) {
      int r = i >> 3, c4 = (i & 7) * 4;
      float f2v = sF2[r * COMP + p];
      float4 f1v = *(const float4*)&sF1[r * COMP + qb + c4];
      pk4 w;
      w.h[0] = (bf16_t)(f2v * f1v.x); w.h[1] = (bf16_t)(f2v * f1v.y);
      w.h[2] = (bf16_t)(f2v * f1v.z); w.h[3] = (bf16_t)(f2v * f1v.w);
      *(uint2*)&sA[r * LDSS + c4] = w.u;
    }
    { // Wf1T tile: b128 copy
      int n = tid >> 2, seg = (tid & 3) * 8;
      *(v4u*)&sBT[n * LDSS + seg] =
          *(const v4u*)&Wf1T[(size_t)n * (COMP * COMP) + kc * KC + seg];
    }
    __syncthreads();
    v16bf a = frag_a(sA + mt * 16 * LDSS);
    #pragma unroll
    for (int j = 0; j < 2; ++j) {
      v16bf b = frag_b(sBT + (nt0 + j) * 16 * LDSS);
      acc[j] = wmma_bf16(a, b, acc[j]);
    }
    __syncthreads();
  }

  // h tile -> LDS, add bf1
  const int lane = tid & 31, lm = lane & 15, lh = lane >> 4;
  #pragma unroll
  for (int j = 0; j < 2; ++j) {
    const int col = (nt0 + j) * 16 + lm;
    const float bv = bf1[col];
    #pragma unroll
    for (int r = 0; r < 8; ++r) {
      const int row = mt * 16 + lh * 8 + r;
      sH[row * 66 + col] = acc[j][r] + bv;
    }
  }
  __syncthreads();

  // tiny head: out = h @ Wf2 + bf2 (64 -> 2), one thread per row
  if (tid < ROWS2) {
    float s0 = bf2[0], s1 = bf2[1];
    #pragma unroll 8
    for (int j = 0; j < COMP; ++j) {
      float h = sH[tid * 66 + j];
      s0 += h * Wf2[j * 2 + 0];
      s1 += h * Wf2[j * 2 + 1];
    }
    Out[(size_t)(row0 + tid) * 2 + 0] = s0;
    Out[(size_t)(row0 + tid) * 2 + 1] = s1;
  }
}

extern "C" void kernel_launch(void* const* d_in, const int* in_sizes, int n_in,
                              void* d_out, int out_size, void* d_ws, size_t ws_size,
                              hipStream_t stream) {
  const float* x1  = (const float*)d_in[0];
  const float* x2  = (const float*)d_in[1];
  const float* Wb1 = (const float*)d_in[2];
  const float* bb1 = (const float*)d_in[3];
  const float* Wb2 = (const float*)d_in[4];
  const float* bb2 = (const float*)d_in[5];
  const float* Wc1 = (const float*)d_in[6];
  const float* bc1 = (const float*)d_in[7];
  const float* Wc2 = (const float*)d_in[8];
  const float* bc2 = (const float*)d_in[9];
  const float* Wf1 = (const float*)d_in[10];
  const float* bf1 = (const float*)d_in[11];
  const float* Wf2 = (const float*)d_in[12];
  const float* bf2 = (const float*)d_in[13];
  float* out = (float*)d_out;

  char* ws = (char*)d_ws;
  size_t off = 0;
  bf16_t* h1 = (bf16_t*)(ws + off); off += (size_t)B_TOTAL * HID * 2;
  bf16_t* h2 = (bf16_t*)(ws + off); off += (size_t)B_TOTAL * HID * 2;
  float*  f1 = (float*) (ws + off); off += (size_t)B_TOTAL * COMP * 4;
  float*  f2 = (float*) (ws + off); off += (size_t)B_TOTAL * COMP * 4;
  bf16_t* WbT1 = (bf16_t*)(ws + off); off += (size_t)DIN * HID * 2;
  bf16_t* WbT2 = (bf16_t*)(ws + off); off += (size_t)DIN * HID * 2;
  bf16_t* WcT1 = (bf16_t*)(ws + off); off += (size_t)HID * COMP * 2;
  bf16_t* WcT2 = (bf16_t*)(ws + off); off += (size_t)HID * COMP * 2;
  bf16_t* Wf1T = (bf16_t*)(ws + off); off += (size_t)COMP * COMP * COMP * 2;

  dim3 blk(256, 1, 1);

  // weight prep: convert + transpose once
  k_cvt_t<<<dim3(DIN / 64, HID / 64), blk, 0, stream>>>(Wb1, WbT1, DIN, HID);
  k_cvt_t<<<dim3(DIN / 64, HID / 64), blk, 0, stream>>>(Wb2, WbT2, DIN, HID);
  k_cvt_t<<<dim3(HID / 64, COMP / 64), blk, 0, stream>>>(Wc1, WcT1, HID, COMP);
  k_cvt_t<<<dim3(HID / 64, COMP / 64), blk, 0, stream>>>(Wc2, WcT2, HID, COMP);
  k_cvt_t<<<dim3((COMP * COMP) / 64, COMP / 64), blk, 0, stream>>>(
      Wf1, Wf1T, COMP * COMP, COMP);

  k_branch<<<dim3(B_TOTAL / ROWS1, 2), blk, 0, stream>>>(
      x1, WbT1, bb1, h1, x2, WbT2, bb2, h2);
  k_compress<<<dim3(B_TOTAL / ROWS2, 2), blk, 0, stream>>>(
      h1, WcT1, bc1, f1, h2, WcT2, bc2, f2);
  k_kron_head<<<dim3(B_TOTAL / ROWS2, 1), blk, 0, stream>>>(
      f1, f2, Wf1T, bf1, Wf2, bf2, out);
}
